// Quant3Linear_29953101922804
// MI455X (gfx1250) — matrix operand order repaired and checked by reading, not compiled, not run on hardware
//
#include <hip/hip_runtime.h>

// y = x @ W^T + bias; W = 4-bit K-means codes (2/byte) + per-row 16-entry codebook.
// 16 output rows per wave = WMMA N dim. Dequant via a per-row 256-entry LDS LUT of
// PRE-PACKED half2 pairs (one ds_load_b32 yields two K values already in B-fragment
// form). A = x broadcast to all 16 M rows, so D row 0 (acc[0], lanes 0..15) is y.

typedef _Float16 v16h __attribute__((ext_vector_type(16)));
typedef _Float16 h2   __attribute__((ext_vector_type(2)));
typedef float    v8f  __attribute__((ext_vector_type(8)));

#define IN_FEATURES    8192
#define OUT_FEATURES   28672
#define ROWS_PER_BLOCK 32        // 2 waves * 16 rows
#define PSTRIDE        257       // pair-table stride (uints): (257r+b)%64 = (r+b)%64

union ABU { unsigned int u[8]; v16h v; };

__global__ __launch_bounds__(64) void qlinear_wmma_kernel(
    const float* __restrict__ x,
    const unsigned char* __restrict__ qweight,
    const float* __restrict__ kmvalues,
    const float* __restrict__ bias,
    float* __restrict__ out)
{
    __shared__ unsigned int   xh2[IN_FEATURES / 2];            // x as packed half2 (16 KB)
    __shared__ unsigned int   pairt[ROWS_PER_BLOCK * PSTRIDE]; // packed-pair LUTs (32.1 KB)
    __shared__ unsigned short tsm16[ROWS_PER_BLOCK * 16];      // f16 codebook bits (1 KB)

    const int tid  = threadIdx.x;
    const int wave = tid >> 5;
    const int lane = tid & 31;
    const int ncol = lane & 15;   // WMMA column / row-in-wave
    const int half = lane >> 4;   // which K half-window this lane feeds
    const int row0 = blockIdx.x * ROWS_PER_BLOCK;

    // ---- phase 1: stage x (f32 -> packed f16 pairs) and the raw f16 codebooks ----
    for (int i = tid; i < IN_FEATURES / 2; i += 64) {
        float2 xf = reinterpret_cast<const float2*>(x)[i];
#if __has_builtin(__builtin_amdgcn_cvt_pkrtz)
        xh2[i] = __builtin_bit_cast(unsigned int,
                     __builtin_amdgcn_cvt_pkrtz(xf.x, xf.y));  // v_cvt_pk_rtz_f16_f32
#else
        h2 p = { (_Float16)xf.x, (_Float16)xf.y };
        xh2[i] = __builtin_bit_cast(unsigned int, p);
#endif
    }
    for (int i = tid; i < ROWS_PER_BLOCK * 16; i += 64) {
        _Float16 v = (_Float16)kmvalues[(size_t)row0 * 16 + i];
        tsm16[i] = __builtin_bit_cast(unsigned short, v);
    }
    __syncthreads();

    // ---- phase 2: expand each row's 16-entry codebook into a 256-entry packed-
    //      pair LUT: pair[byte] = T[lo] | T[hi]<<16  (= half2{even K, odd K}) ----
    for (int i = tid; i < ROWS_PER_BLOCK * 256; i += 64) {
        const int r = i >> 8;
        const unsigned int byt = i & 255;
        unsigned int lo = tsm16[r * 16 + (byt & 15u)];
        unsigned int hi = tsm16[r * 16 + (byt >> 4)];
        pairt[r * PSTRIDE + byt] = lo | (hi << 16);
    }
    __syncthreads();

    const int row = row0 + wave * 16 + ncol;          // lanes L and L+16 share a row
    const uint2* __restrict__ q64 =
        reinterpret_cast<const uint2*>(qweight + (size_t)row * (IN_FEATURES / 2));
    const int tb = (wave * 16 + ncol) * PSTRIDE;      // this lane's pair-LUT base

    v8f acc = {};                                      // C/D accumulator (f32 16x16)

    for (int co = 0; co < 16; ++co) {                 // 16 x 16 = 256 WMMAs, K=32 each
        if (co < 8)                                   // stay inside this row (4 KB)
            __builtin_prefetch(reinterpret_cast<const char*>(q64) + co * 256 + 2048,
                               0, 3);                 // near-scope global_prefetch_b8
#pragma unroll 4
        for (int ci = 0; ci < 16; ++ci) {
            const int c = co * 16 + ci;
            // 8 bytes of codes: K = [c*32 + 16*half, +16) of this lane's row.
            uint2 w = q64[c * 2 + half];

            // ---- B fragment (32x16 f16): VGPR j = packed K {2j+16h, 2j+16h+1}
            //      = pre-packed pair for byte (j + 8h): bfe + lshl_add + ds_load_b32
            ABU b;
#pragma unroll
            for (int j = 0; j < 8; ++j) {
                unsigned int word = (j < 4) ? w.x : w.y;
                unsigned int byt  = (word >> (8 * (j & 3))) & 0xFFu;
                b.u[j] = pairt[tb + byt];
            }

            // ---- A fragment (16x32 f16): x broadcast to all 16 M rows.
            //      pair idx: j<4 -> j+8h ; j>=4 -> 4+(j-4)+8h  (two ds_load_b128)
            ABU a;
            const int kb = c * 16;
#pragma unroll
            for (int j = 0; j < 4; ++j) {
                a.u[j]     = xh2[kb + j + 8 * half];
                a.u[j + 4] = xh2[kb + 4 + j + 8 * half];
            }

            acc = __builtin_amdgcn_wmma_f32_16x16x32_f16(
                false, a.v, false, b.v, (short)0, acc, false, false);
        }
    }

    // D row M=0 lives in acc[0], lanes 0..15, N = lane. All M rows identical.
    if (lane < 16)
        out[row] = acc[0] + bias[row];
}

extern "C" void kernel_launch(void* const* d_in, const int* in_sizes, int n_in,
                              void* d_out, int out_size, void* d_ws, size_t ws_size,
                              hipStream_t stream) {
    const float*         x    = (const float*)d_in[0];
    const unsigned char* qw   = (const unsigned char*)d_in[1];
    const float*         kmv  = (const float*)d_in[2];
    const float*         bias = (const float*)d_in[3];
    float*               out  = (float*)d_out;

    dim3 grid(OUT_FEATURES / ROWS_PER_BLOCK);   // 896 blocks
    dim3 block(64);                             // 2 wave32
    hipLaunchKernelGGL(qlinear_wmma_kernel, grid, block, 0, stream,
                       x, qw, kmv, bias, out);
}
